// Lattice_88751204205226
// MI455X (gfx1250) — compile-verified, matrix operand order checked by reading, not run on hardware
//
#include <hip/hip_runtime.h>
#include <math.h>

// Problem dims (match reference)
#define Hh   512
#define Bb   128
#define Tt   512
#define Ll   4
#define Jj   2
#define H2   1024           // 2*H (GEMM K)
#define H4   2048           // 4*H (lin N)
#define KTILES 32           // K / 32
#define TILE_ELT (KTILES * 512)  // packed elements per 16-col n-tile
#define RDEPTH 8            // h0 pipeline ring depth
#define GWG  8              // workgroups per layer group (== 16-row M tiles)
#define BH   (Bb * Hh)

typedef __bf16 bf16_t;
typedef bf16_t v16bf __attribute__((ext_vector_type(16)));
typedef bf16_t v8bf  __attribute__((ext_vector_type(8)));
typedef float  v8f   __attribute__((ext_vector_type(8)));
typedef int    v4i   __attribute__((vector_size(16)));

#if __has_builtin(__builtin_amdgcn_global_load_async_to_lds_b128) && \
    __has_builtin(__builtin_amdgcn_s_wait_asynccnt)
#define HAVE_ASYNC_LDS 1
#else
#define HAVE_ASYNC_LDS 0
#endif

// 16-byte global->LDS copy: async DMA on gfx1250 when available.
__device__ __forceinline__ void cp16(bf16_t* dstL, const bf16_t* srcG) {
#if HAVE_ASYNC_LDS
  typedef __attribute__((address_space(1))) v4i as1_v4i;
  typedef __attribute__((address_space(3))) v4i as3_v4i;
  __builtin_amdgcn_global_load_async_to_lds_b128(
      (as1_v4i*)(unsigned long long)srcG,
      (as3_v4i*)(unsigned long long)dstL, 0, 0);
#else
  *(uint4*)dstL = *(const uint4*)srcG;
#endif
}
__device__ __forceinline__ void cp16_wait() {
#if HAVE_ASYNC_LDS
  __builtin_amdgcn_s_wait_asynccnt(0);
#endif
}

// ---- fast transcendentals ---------------------------------------------------
__device__ __forceinline__ float fast_rcp(float x) {
#if __has_builtin(__builtin_amdgcn_rcpf)
  return __builtin_amdgcn_rcpf(x);        // v_rcp_f32
#else
  return 1.0f / x;
#endif
}
__device__ __forceinline__ float sigmoidf_(float x) {
  return fast_rcp(1.0f + __expf(-x));
}
__device__ __forceinline__ float tanhf_(float x) {
#if __has_builtin(__builtin_amdgcn_tanhf)
  return __builtin_amdgcn_tanhf(x);       // gfx1250 v_tanh_f32
#else
  float e = __expf(2.0f * x);
  return (e - 1.0f) * fast_rcp(e + 1.0f);
#endif
}

// ---------------------------------------------------------------------------
// Weight pre-pack: fp32 (K2 x N row major) -> bf16 WMMA B-fragment tile order.
// [mat][ntile][ktile][lane(32)][16 bf16]; lane=16*khalf+col,
// element i -> W[kk*32 + khalf*16 + i][nt*16 + col]
// ---------------------------------------------------------------------------
__global__ void pack_weights_kernel(const float* __restrict__ src,
                                    bf16_t* __restrict__ dst,
                                    int K2, int N, int nMat) {
  const long perMat = (long)K2 * N;
  const long total  = perMat * nMat;
  const int  kTiles = K2 / 32;
  const long stride = (long)gridDim.x * blockDim.x;
  for (long idx = (long)blockIdx.x * blockDim.x + threadIdx.x; idx < total;
       idx += stride) {
    long mat  = idx / perMat;
    long rem  = idx - mat * perMat;
    int  tile = (int)(rem >> 9);
    int  p    = (int)(rem & 511);
    int  nt   = tile / kTiles;
    int  kk   = tile - nt * kTiles;
    int  lane = p >> 4;
    int  i    = p & 15;
    int  k    = kk * 32 + (lane >> 4) * 16 + i;
    int  n    = nt * 16 + (lane & 15);
    dst[idx] = (bf16_t)src[mat * perMat + (long)k * N + n];
  }
}

// Zero recurrent state (bf16, as ints) + flags each launch.
__global__ void init_ws_kernel(int* h1zero, int nInts, int* flags, int nFlags) {
  int i = blockIdx.x * blockDim.x + threadIdx.x;
  int stride = gridDim.x * blockDim.x;
  for (int k = i; k < nInts; k += stride) h1zero[k] = 0;
  for (int k = i; k < nFlags; k += stride) flags[k] = 0;
}

// ---------------------------------------------------------------------------
// 4 n-tiles per A-fragment load, with a *split* A matrix: k in [0,512) comes
// from region P (row stride sP), k in [512,1024) from region Q (row stride sQ).
// ---------------------------------------------------------------------------
__device__ __forceinline__ void wmma_tile4_dual(const bf16_t* aP, int sP,
                                                const bf16_t* aQ, int sQ,
                                                const bf16_t* __restrict__ Wt,
                                                int lane, v8f acc[4]) {
  const bf16_t* ap = aP + (lane & 15) * sP + (lane >> 4) * 16;
  const bf16_t* aq = aQ + (lane & 15) * sQ + (lane >> 4) * 16;
  const bf16_t* bptr = Wt + lane * 16;
#pragma unroll 2
  for (int kk = 0; kk < 16; ++kk) {
    v16bf a = *(const v16bf*)(ap + kk * 32);
#pragma unroll
    for (int m = 0; m < 4; ++m) {
      v16bf b = *(const v16bf*)(bptr + kk * 512 + m * TILE_ELT);
      acc[m] = __builtin_amdgcn_wmma_f32_16x16x32_bf16(
          false, a, false, b, (short)0, acc[m], false, false);
    }
  }
#pragma unroll 2
  for (int kk = 16; kk < 32; ++kk) {
    v16bf a = *(const v16bf*)(aq + (kk - 16) * 32);
#pragma unroll
    for (int m = 0; m < 4; ++m) {
      v16bf b = *(const v16bf*)(bptr + kk * 512 + m * TILE_ELT);
      acc[m] = __builtin_amdgcn_wmma_f32_16x16x32_bf16(
          false, a, false, b, (short)0, acc[m], false, false);
    }
  }
}

// ---------------------------------------------------------------------------
// Persistent lattice kernel. Grid = Ll * GWG blocks of 256 threads (8 wave32).
// Layers pipelined across timesteps; bf16 h0 slices flow l -> l+1 through an
// L2-resident ring with pairwise produced/consumed flags. All intra-cell
// phase-to-phase traffic (gates g, concat halves, products) lives in LDS:
// 32 KB sA + 32 KB sR + 64 KB sG = 128 KB of the WGP's 320 KB.
// ---------------------------------------------------------------------------
__global__ __launch_bounds__(256, 1)
void lattice_persistent_kernel(const float* __restrict__ H0in,
                               const float* __restrict__ lin_b,
                               const float* __restrict__ g0_b,
                               const float* __restrict__ g1_b,
                               const bf16_t* __restrict__ Wlin,
                               const bf16_t* __restrict__ Wg0,
                               const bf16_t* __restrict__ Wg1,
                               bf16_t* __restrict__ h0loc_all,
                               bf16_t* __restrict__ h1buf_all,
                               bf16_t* __restrict__ pipe,
                               int* __restrict__ flags,
                               float* __restrict__ out) {
  __shared__ __attribute__((aligned(128))) bf16_t sA[16 * H2];      // 32 KB [h0|h1]
  __shared__ __attribute__((aligned(128))) bf16_t sR[2 * 16 * Hh];  // 32 KB h1*r | h0*q
  __shared__ __attribute__((aligned(128))) bf16_t sG[16 * H4];      // 64 KB gates

  const int bid   = blockIdx.x;
  const int l     = bid / GWG;
  const int wg    = bid - l * GWG;
  const int tid   = threadIdx.x;
  const int wave  = tid >> 5;
  const int lane  = tid & 31;
  const int mrow0 = wg * 16;

  bf16_t* h0loc = h0loc_all + (long)l * BH;
  bf16_t* h1b0  = h1buf_all + (long)(l * 2 + 0) * BH;
  bf16_t* h1b1  = h1buf_all + (long)(l * 2 + 1) * BH;

  int* prodF = flags;             // [Ll-1][GWG]
  int* consF = flags + 3 * GWG;   // [Ll-1][GWG]

  for (int t = 0; t < Tt; ++t) {
    if (tid == 0) {
      if (l > 0) {
        int* f = &prodF[(l - 1) * GWG + wg];
        while (__hip_atomic_load(f, __ATOMIC_ACQUIRE,
                                 __HIP_MEMORY_SCOPE_AGENT) <= t)
          __builtin_amdgcn_s_sleep(1);
      }
      if (l < Ll - 1) {
        int* f = &consF[l * GWG + wg];
        while (__hip_atomic_load(f, __ATOMIC_ACQUIRE,
                                 __HIP_MEMORY_SCOPE_AGENT) < t - RDEPTH + 1)
          __builtin_amdgcn_s_sleep(1);
      }
    }
    __syncthreads();
    __builtin_amdgcn_fence(__ATOMIC_ACQUIRE, "agent");

    for (int j = 0; j < Jj; ++j) {
      const int cell = l * Jj + j;
      const int cidx = t * 2 + j;
      const bf16_t* h1r = (cidx & 1) ? h1b1 : h1b0;
      bf16_t*       h1w = (cidx & 1) ? h1b0 : h1b1;

      const bf16_t* h0p;   // bf16 source (unused for l==0 && j==0)
      if (j == 0)
        h0p = (l == 0) ? (const bf16_t*)nullptr
            : pipe + ((long)(l - 1) * RDEPTH + (t & (RDEPTH - 1))) * BH;
      else
        h0p = h0loc;

      const bf16_t* WlinC = Wlin + (long)cell * ((long)H2 * H4);
      const bf16_t* Wg0C  = Wg0  + (long)cell * ((long)H2 * Hh);
      const bf16_t* Wg1C  = Wg1  + (long)cell * ((long)H2 * Hh);
      const float*  linbC = lin_b + (long)cell * H4;
      const float*  g0bC  = g0_b  + (long)cell * Hh;
      const float*  g1bC  = g1_b  + (long)cell * Hh;

      // ---- stage A = [h0 | h1] into sA (16-byte chunks) -----------------
      if (l == 0 && j == 0) {
        for (int ch = tid; ch < 16 * 64; ch += 256) {   // left: fp32 input
          int r = ch >> 6, c0 = (ch & 63) * 8;
          const float* s =
              H0in + (long)t * Hh + (long)(mrow0 + r) * ((long)Tt * Hh) + c0;
          float4 f0 = *(const float4*)s;
          float4 f1 = *(const float4*)(s + 4);
          bf16_t tmp[8] = {(bf16_t)f0.x, (bf16_t)f0.y, (bf16_t)f0.z,
                           (bf16_t)f0.w, (bf16_t)f1.x, (bf16_t)f1.y,
                           (bf16_t)f1.z, (bf16_t)f1.w};
          *(uint4*)&sA[r * H2 + c0] = *(const uint4*)tmp;
        }
        for (int ch = tid; ch < 16 * 64; ch += 256) {   // right: h1 copy
          int r = ch >> 6, c0 = (ch & 63) * 8;
          cp16(&sA[r * H2 + Hh + c0], h1r + (long)(mrow0 + r) * Hh + c0);
        }
      } else {
        for (int ch = tid; ch < 16 * 64; ch += 256) {
          int r = ch >> 6, c0 = (ch & 63) * 8;
          int b = mrow0 + r;
          cp16(&sA[r * H2 + c0],      h0p + (long)b * Hh + c0);
          cp16(&sA[r * H2 + Hh + c0], h1r + (long)b * Hh + c0);
        }
      }
      cp16_wait();
      __syncthreads();

      // ---- phase 1: g = sigmoid([h0,h1] @ lin_w + lin_b) -> sG (bf16) ---
#pragma unroll 1
      for (int ib = 0; ib < 4; ++ib) {        // 16 tiles/wave, 4 at a time
        int nt0 = wave * 16 + ib * 4;
        const bf16_t* Wt = WlinC + (long)nt0 * TILE_ELT;
        __builtin_prefetch(Wt + 4 * TILE_ELT, 0, 1);
        v8f acc[4] = {};
        wmma_tile4_dual(sA, H2, sA + Hh, H2, Wt, lane, acc);
#pragma unroll
        for (int m = 0; m < 4; ++m) {
          int col = (nt0 + m) * 16 + (lane & 15);
          float bias = linbC[col];
#pragma unroll
          for (int v = 0; v < 8; ++v) {
            int lr = (lane >> 4) * 8 + v;     // local row 0..15
            sG[lr * H4 + col] = (bf16_t)sigmoidf_(acc[m][v] + bias);
          }
        }
      }
      __syncthreads();

      // ---- phase 2 staging: elementwise products via packed bf16 --------
      // sR[0] = h1*r (A0 right half); sR[16*Hh] = h0*q (A1 right half)
      for (int ch = tid; ch < 16 * 64; ch += 256) {
        int r = ch >> 6, c0 = (ch & 63) * 8;
        v8bf h0v = *(const v8bf*)&sA[r * H2 + c0];
        v8bf h1v = *(const v8bf*)&sA[r * H2 + Hh + c0];
        v8bf rr  = *(const v8bf*)&sG[r * H4 + 2 * Hh + c0];
        v8bf qq  = *(const v8bf*)&sG[r * H4 + 3 * Hh + c0];
        *(v8bf*)&sR[r * Hh + c0]           = h1v * rr;   // v_pk_mul_bf16
        *(v8bf*)&sR[16 * Hh + r * Hh + c0] = h0v * qq;
      }
      __syncthreads();

      const bool dst32 = (j == 1) && (l == Ll - 1);
      bf16_t* h0d16 = (j == 0)
          ? h0loc
          : pipe + ((long)l * RDEPTH + (t & (RDEPTH - 1))) * BH;

      // ---- phase 2 GEMMs + gating epilogue ------------------------------
#pragma unroll 1
      for (int ib = 0; ib < 2; ++ib) {        // 8 tiles/wave, 4 at a time
        int u0 = wave * 8 + ib * 4;           // batch entirely in g0 or g1
        bool isG1 = (u0 >= 32);
        int nt0 = isG1 ? (u0 - 32) : u0;
        const bf16_t* W = (isG1 ? Wg1C : Wg0C) + (long)nt0 * TILE_ELT;
        __builtin_prefetch(W + 4 * TILE_ELT, 0, 1);
        v8f acc[4] = {};
        if (!isG1)   // A0 = [h0 | h1*r]
          wmma_tile4_dual(sA, H2, sR, Hh, W, lane, acc);
        else         // A1 = [h1 | h0*q]
          wmma_tile4_dual(sA + Hh, H2, sR + 16 * Hh, Hh, W, lane, acc);
#pragma unroll
        for (int m = 0; m < 4; ++m) {
          int col = (nt0 + m) * 16 + (lane & 15);
          float bias = (isG1 ? g1bC : g0bC)[col];
#pragma unroll
          for (int v = 0; v < 8; ++v) {
            int lr = (lane >> 4) * 8 + v;     // local row 0..15
            int b  = mrow0 + lr;
            float cap = tanhf_(acc[m][v] + bias);
            if (!isG1) {  // h0_cap -> h1_new = z1*cap + (1-z1)*h1
              float z1  = (float)sG[lr * H4 + Hh + col];
              float h1o = (float)sA[lr * H2 + Hh + col];
              h1w[(long)b * Hh + col] = (bf16_t)(z1 * cap + (1.0f - z1) * h1o);
            } else {      // h1_cap -> h0_new = z0*cap + (1-z0)*h0
              float z0  = (float)sG[lr * H4 + col];
              float h0o = (float)sA[lr * H2 + col];
              float h0n = z0 * cap + (1.0f - z0) * h0o;
              if (dst32) out[(long)b * Tt * Hh + (long)t * Hh + col] = h0n;
              else       h0d16[(long)b * Hh + col] = (bf16_t)h0n;
            }
          }
        }
      }
      __syncthreads();

      if (tid == 0) {
        if (j == 0 && l > 0) {
          __builtin_amdgcn_fence(__ATOMIC_RELEASE, "agent");
          __hip_atomic_fetch_add(&consF[(l - 1) * GWG + wg], 1,
                                 __ATOMIC_RELEASE, __HIP_MEMORY_SCOPE_AGENT);
        }
        if (j == 1 && l < Ll - 1) {
          __builtin_amdgcn_fence(__ATOMIC_RELEASE, "agent");
          __hip_atomic_store(&prodF[l * GWG + wg], t + 1,
                             __ATOMIC_RELEASE, __HIP_MEMORY_SCOPE_AGENT);
        }
      }
      __syncthreads();
    }
  }

  // H1_final: 2*T cells toggled the double buffer back to index 0.
  const bf16_t* h1f = h1b0;
  for (int e = tid; e < 16 * Hh; e += 256) {
    int r = e >> 9, col = e & (Hh - 1);
    int b = mrow0 + r;
    out[(long)Bb * Tt * Hh + ((long)b * Ll + l) * Hh + col] =
        (float)h1f[(long)b * Hh + col];
  }
}

// ---------------------------------------------------------------------------
extern "C" void kernel_launch(void* const* d_in, const int* in_sizes, int n_in,
                              void* d_out, int out_size, void* d_ws,
                              size_t ws_size, hipStream_t stream) {
  (void)in_sizes; (void)n_in; (void)out_size; (void)ws_size;
  const float* H0in  = (const float*)d_in[0];
  const float* lin_w = (const float*)d_in[1];
  const float* lin_b = (const float*)d_in[2];
  const float* g0_w  = (const float*)d_in[3];
  const float* g0_b  = (const float*)d_in[4];
  const float* g1_w  = (const float*)d_in[5];
  const float* g1_b  = (const float*)d_in[6];
  float* out = (float*)d_out;

  char* ws = (char*)d_ws;
  size_t off = 0;
  bf16_t* Wlin = (bf16_t*)(ws + off); off += (size_t)Ll * Jj * H2 * H4 * 2;
  bf16_t* Wg0  = (bf16_t*)(ws + off); off += (size_t)Ll * Jj * H2 * Hh * 2;
  bf16_t* Wg1  = (bf16_t*)(ws + off); off += (size_t)Ll * Jj * H2 * Hh * 2;
  bf16_t* h0loc = (bf16_t*)(ws + off); off += (size_t)Ll * BH * 2;
  bf16_t* h1buf = (bf16_t*)(ws + off); off += (size_t)Ll * 2 * BH * 2;
  bf16_t* pipe  = (bf16_t*)(ws + off); off += (size_t)(Ll - 1) * RDEPTH * BH * 2;
  int*   flags = (int*)(ws + off);    off += 256;

  init_ws_kernel<<<256, 256, 0, stream>>>((int*)h1buf, Ll * BH, flags, 64);
  pack_weights_kernel<<<2048, 256, 0, stream>>>(lin_w, Wlin, H2, H4, Ll * Jj);
  pack_weights_kernel<<<1024, 256, 0, stream>>>(g0_w, Wg0, H2, Hh, Ll * Jj);
  pack_weights_kernel<<<1024, 256, 0, stream>>>(g1_w, Wg1, H2, Hh, Ll * Jj);
  lattice_persistent_kernel<<<Ll * GWG, 256, 0, stream>>>(
      H0in, lin_b, g0_b, g1_b, Wlin, Wg0, Wg1,
      h0loc, h1buf, pipe, flags, out);
}
